// ConvMosaic_12910671692231
// MI455X (gfx1250) — matrix operand order repaired and checked by reading, not compiled, not run on hardware
//
#include <hip/hip_runtime.h>

// ---------------------------------------------------------------------------
// ConvMosaic for MI455X (gfx1250): phase-decomposed im2col GEMM on WMMA.
//   x      : (4, 32, 256, 256) f32
//   weight : (16, 288, 64)     f32   -- phase s = w % 16 selects weight[s]
//   out    : (4, 64, 256, 256) f32
// Strategy: split-bf16 (hi+lo) WMMA f32_16x16x32_bf16, f32 accumulate.
// Round 2: full k-chunk unroll (folds div/address math), v_perm_b32 repack,
//          non-temporal output stores.
// ---------------------------------------------------------------------------

typedef __attribute__((ext_vector_type(16))) __bf16 v16bf;
typedef __attribute__((ext_vector_type(8)))  float  v8f;

#define NCH    32
#define HDIM   256
#define WDIM   256
#define OUTC   64
#define KDIM   288
#define NPH    16          // mosaic phases
#define HB     4           // h-rows per workgroup
#define LROWS  (HB + 2)    // staged input rows (halo)
#define LCOLS  (WDIM + 2)  // staged cols (halo)
#define LSZ    (NCH * LROWS * LCOLS)          // 49536 u32 = 198144 B LDS
#define WFRAG_U32 (NPH * 9 * 4 * 32 * 8)      // 147456 u32 per (hi|lo) plane

__device__ __forceinline__ unsigned f2bf(float f) {
  unsigned u = __float_as_uint(f);
  u += 0x7FFFu + ((u >> 16) & 1u);            // round-to-nearest-even
  return u >> 16;
}
__device__ __forceinline__ float bf2f(unsigned h) {
  return __uint_as_float(h << 16);
}

// ---------------------------------------------------------------------------
// Kernel 0: convert weights f32 -> (bf16_hi, bf16_lo), pre-swizzled into the
// CDNA5 16-bit B-matrix (32x16) per-lane layout, fragment-linear:
//   index = ((((s*9 + t)*4 + j)*32 + lane)*8 + v)
// u32 packs the two 16-bit K-halves (e=0 low, e=1 high) of VGPR v.
// B layout: K = (v>=4?16:0) + (lane>=16?8:0) + (v&3)*2 + e, matching the A fragments.
// ---------------------------------------------------------------------------
__global__ void mosaic_prepack(const float* __restrict__ wt,
                               unsigned* __restrict__ whi,
                               unsigned* __restrict__ wlo) {
  int idx = blockIdx.x * 256 + threadIdx.x;
  if (idx >= WFRAG_U32) return;
  int v  = idx & 7;
  int l  = (idx >> 3) & 31;
  int j  = (idx >> 8) & 3;
  int st = idx >> 10;          // s*9 + t
  int t  = st % 9;
  int s  = st / 9;
  int n  = (j << 4) + (l & 15);
  unsigned hi2 = 0, lo2 = 0;
  #pragma unroll
  for (int e = 0; e < 2; ++e) {
    int kk = ((v >> 2) << 4) + ((l >= 16) ? 8 : 0) + ((v & 3) << 1) + e;
    int k  = (t << 5) + kk;
    float f = wt[(s * KDIM + k) * OUTC + n];
    unsigned hi = f2bf(f);
    unsigned lo = f2bf(f - bf2f(hi));
    hi2 |= hi << (16 * e);
    lo2 |= lo << (16 * e);
  }
  whi[idx] = hi2;
  wlo[idx] = lo2;
}

// ---------------------------------------------------------------------------
// Kernel 1: GEMM. Grid = N * (H/HB) blocks of 512 threads (16 waves).
// Wave w handles phase p = w: M = HB x 16 pixels (4 M-tiles), N = 64 (4 tiles).
// ---------------------------------------------------------------------------
__global__ __launch_bounds__(512) void mosaic_gemm(
    const float* __restrict__ x,
    const unsigned* __restrict__ whi,
    const unsigned* __restrict__ wlo,
    float* __restrict__ out) {
  __shared__ unsigned ls[LSZ];

  const int tid    = threadIdx.x;
  const int lane   = tid & 31;
  const int p      = tid >> 5;            // phase = wave id (0..15)
  const int nb     = blockIdx.x >> 6;     // batch
  const int h0     = (blockIdx.x & 63) * HB;
  const int m      = lane & 15;           // M-row within tile
  const int hiHalf = (lane >= 16) ? 1 : 0;
  const int wpix   = p + (m << 4);        // pixel column of this A row

  // ---- Stage x[h0-1 .. h0+HB] into LDS as packed {lo16|hi16} bf16 pairs ----
  for (int idx = tid; idx < LSZ; idx += 512) {
    int c   = idx / (LROWS * LCOLS);
    int rem = idx - c * (LROWS * LCOLS);
    int r   = rem / LCOLS;
    int wc  = rem - r * LCOLS;
    int hh  = h0 - 1 + r;
    int ww  = wc - 1;
    float val = 0.0f;
    if ((unsigned)hh < HDIM && (unsigned)ww < WDIM)
      val = x[((nb * NCH + c) * HDIM + hh) * WDIM + ww];
    unsigned hi = f2bf(val);
    unsigned lo = f2bf(val - bf2f(hi));
    ls[idx] = hi | (lo << 16);
  }
  __syncthreads();

  // ---- K loop: 9 chunks of 32, split-bf16 triple-WMMA accumulate ----------
  v8f acc[HB][4];
  #pragma unroll
  for (int i = 0; i < HB; ++i)
    #pragma unroll
    for (int j = 0; j < 4; ++j)
      acc[i][j] = 0;

  // Fully unrolled: k is compile-time, so /9 and /3 fold and every LDS
  // address becomes wpix + constant (absorbed into DS offsets / CSE'd).
  #pragma unroll
  for (int t = 0; t < 9; ++t) {
    // A fragments (16x32 bf16, ISA 16-bit A layout) for the 4 M-tiles.
    v16bf Ah[HB], Al[HB];
    #pragma unroll
    for (int i = 0; i < HB; ++i) {
      unsigned pa[16];
      #pragma unroll
      for (int vv = 0; vv < 8; ++vv) {
        #pragma unroll
        for (int e = 0; e < 2; ++e) {
          // A layout: K = (v>=4?16:0) + (lane>=16?8:0) + (v&3)*2 + e
          int kk = ((vv >> 2) << 4) + (hiHalf << 3) + ((vv & 3) << 1) + e;
          int k  = 32 * t + kk;
          int c  = k / 9;
          int r9 = k - 9 * c;
          int di = r9 / 3;
          int dj = r9 - 3 * di;
          pa[2 * vv + e] = ls[(c * LROWS + i + di) * LCOLS + (wpix + dj)];
        }
      }
      union { unsigned u[8]; v16bf v; } uh, ul;
      #pragma unroll
      for (int vv = 0; vv < 8; ++vv) {
        unsigned p0 = pa[2 * vv], p1 = pa[2 * vv + 1];
        // byte pool: {p1 bytes 7:4, p0 bytes 3:0}
        uh.u[vv] = __builtin_amdgcn_perm(p1, p0, 0x05040100u);  // {hi0, hi1}
        ul.u[vv] = __builtin_amdgcn_perm(p1, p0, 0x07060302u);  // {lo0, lo1}
      }
      Ah[i] = uh.v;
      Al[i] = ul.v;
    }

    #pragma unroll
    for (int j = 0; j < 4; ++j) {
      // B fragment: coalesced b128 loads from prepacked weights (L2-resident)
      long boff = ((((long)(p * 9 + t) * 4 + j) * 32 + lane) * 8);
      const uint4* bh = (const uint4*)(whi + boff);
      const uint4* bl = (const uint4*)(wlo + boff);
      union { uint4 q[2]; v16bf v; } Bh, Bl;
      Bh.q[0] = bh[0]; Bh.q[1] = bh[1];
      Bl.q[0] = bl[0]; Bl.q[1] = bl[1];
      #pragma unroll
      for (int i = 0; i < HB; ++i) {
        acc[i][j] = __builtin_amdgcn_wmma_f32_16x16x32_bf16(
            false, Ah[i], false, Bh.v, (short)0, acc[i][j], false, false);
        acc[i][j] = __builtin_amdgcn_wmma_f32_16x16x32_bf16(
            false, Ah[i], false, Bl.v, (short)0, acc[i][j], false, false);
        acc[i][j] = __builtin_amdgcn_wmma_f32_16x16x32_bf16(
            false, Al[i], false, Bh.v, (short)0, acc[i][j], false, false);
      }
    }
  }

  // ---- Epilogue: transpose through LDS (stride 257 = conflict-free), then
  //      coalesced non-temporal global stores of out[n, :, h0+i, :] ---------
  float* ols = (float*)ls;   // 64 * 257 * 4B = 65792 B, reuses staging LDS
  #pragma unroll 1
  for (int i = 0; i < HB; ++i) {
    __syncthreads();
    #pragma unroll
    for (int j = 0; j < 4; ++j) {
      #pragma unroll
      for (int r = 0; r < 8; ++r) {
        // C/D layout: VGPR r -> M = r (lanes 0-15) / r+8 (lanes 16-31), N = lane&15
        int mm = r + (hiHalf << 3);
        int oc = (j << 4) + m;
        ols[oc * 257 + (p + (mm << 4))] = acc[i][j][r];
      }
    }
    __syncthreads();
    for (int idx = tid; idx < OUTC * WDIM; idx += 512) {
      int oc = idx >> 8;
      int ww = idx & 255;
      __builtin_nontemporal_store(
          ols[oc * 257 + ww],
          &out[((nb * OUTC + oc) * HDIM + (h0 + i)) * WDIM + ww]);
    }
  }
}

// ---------------------------------------------------------------------------
extern "C" void kernel_launch(void* const* d_in, const int* in_sizes, int n_in,
                              void* d_out, int out_size, void* d_ws, size_t ws_size,
                              hipStream_t stream) {
  const float* x  = (const float*)d_in[0];
  const float* wt = (const float*)d_in[1];
  float* out      = (float*)d_out;
  unsigned* whi   = (unsigned*)d_ws;          // needs 2 * 147456 * 4 B = 1.18 MB
  unsigned* wlo   = whi + WFRAG_U32;

  mosaic_prepack<<<(WFRAG_U32 + 255) / 256, 256, 0, stream>>>(wt, whi, wlo);
  mosaic_gemm<<<dim3(4 * (HDIM / HB)), dim3(512), 0, stream>>>(x, whi, wlo, out);
}
